// ControlFlowGraphEncoder_55490977465024
// MI455X (gfx1250) — matrix operand order repaired and checked by reading, not compiled, not run on hardware
//
#include <hip/hip_runtime.h>

#define DEV __device__ __forceinline__

// ---------------- problem constants ----------------
constexpr int F_IN = 128;
constexpr int H    = 256;
constexpr int HQ   = 64;
constexpr int HH   = 128;
constexpr int GG   = 128;              // graphs
constexpr int LL   = 512;              // nodes per graph
constexpr int NN   = GG * LL;          // 65536 nodes
constexpr int EE   = NN * 8;           // 524288 edges
constexpr int G4   = 4 * HH;           // 512 LSTM gate width

// d_out float offsets (outputs concatenated in return order)
constexpr size_t OFF_GE = (size_t)NN * H;
constexpr size_t OFF_BR = OFF_GE + (size_t)GG * H;
constexpr size_t OFF_LP = OFF_BR + (size_t)GG * 3;
constexpr size_t OFF_XL = OFF_LP + (size_t)GG * 4;

// workspace section sizes (bytes)
constexpr size_t SZ_X   = (size_t)NN * H * 4;    //  67.1 MB f32 activations
constexpr size_t SZ_PRE = (size_t)NN * G4 * 4;   // 134.2 MB gate precompute (overlaps agg bufs)
constexpr size_t SZ_XBF = (size_t)NN * H * 2;    //  33.6 MB bf16 activations

typedef __bf16 v16bf __attribute__((ext_vector_type(16)));
typedef float  v8f   __attribute__((ext_vector_type(8)));

union FragBF { v16bf v; unsigned int u[8]; };

DEV unsigned short bf16r(float f) {           // RNE float -> bf16 bits
  unsigned int u = __builtin_bit_cast(unsigned int, f);
  u += 0x7fffu + ((u >> 16) & 1u);
  return (unsigned short)(u >> 16);
}

DEV v8f wmma_bf16(v16bf a, v16bf b, v8f c) {
  return __builtin_amdgcn_wmma_f32_16x16x32_bf16(false, a, false, b, (short)0, c,
                                                 false, false);
}

// A fragment: 16x32 bf16 tile, row-major source (ld in elems). row = lane&15.
// K layout per ISA 7.12.2: lanes0-15 -> K {0..7,16..23}; lanes16-31 -> K {8..15,24..31}.
DEV v16bf load_frag_a(const unsigned short* base, int ld, int lane) {
  FragBF f;
  const unsigned short* p = base + (lane & 15) * ld + ((lane >> 4) << 3);
#pragma unroll
  for (int v = 0; v < 8; ++v) {
    int k0 = ((v & 4) << 2) + ((v & 3) << 1);   // (v>=4?16:0) + 2*(v&3)
    f.u[v] = *(const unsigned int*)(p + k0);
  }
  return f.v;
}

// B fragment for X @ W^T. W row-major [outF][K]; base = W + n0*ld + kbase.
// B[k][n] = W[n0+n][kbase+k]; lane holds n = lane&15, k = (lane>>4)*8 + (v>=4?16:0) + 2*(v&3) + j.
DEV v16bf load_frag_b(const unsigned short* base, int ld, int lane) {
  FragBF f;
  const unsigned short* p = base + (lane & 15) * ld + ((lane >> 4) << 3);
#pragma unroll
  for (int v = 0; v < 8; ++v)
    f.u[v] = *(const unsigned int*)(p + ((v & 4) << 2) + ((v & 3) << 1));
  return f.v;
}

DEV void atomAddF(float* p, float v) { unsafeAtomicAdd(p, v); }
DEV float sigm(float x) { return 1.0f / (1.0f + expf(-x)); }
DEV float gelu_exact(float x) { return 0.5f * x * (1.0f + erff(x * 0.70710678118654752f)); }

// CDNA5 async global->LDS copy (ASYNCcnt-tracked, §15.18.3 op 98).
// lds_off = byte offset within workgroup LDS (= low 32 bits of generic shared addr).
DEV void async_ld_b128(unsigned lds_off, const void* gaddr) {
  asm volatile("global_load_async_to_lds_b128 %0, %1, off"
               :: "v"(lds_off), "v"((unsigned long long)gaddr) : "memory");
}

// ---------------- utility kernels ----------------
__global__ void k_f32_to_bf16(const float* __restrict__ s, unsigned short* __restrict__ d, int n) {
  int i = blockIdx.x * 256 + threadIdx.x;
  if (i < n) d[i] = bf16r(s[i]);
}

__global__ void k_zero(float* __restrict__ p, int n) {
  int i = blockIdx.x * 256 + threadIdx.x;
  if (i < n) p[i] = 0.0f;
}

__global__ void k_degree(const int* __restrict__ edge, float* __restrict__ cnt) {
  int e = blockIdx.x * 256 + threadIdx.x;
  if (e < EE) atomAddF(&cnt[edge[EE + e]], 1.0f);
}

__global__ void __launch_bounds__(256)
k_aggregate(const int* __restrict__ edge, const float* __restrict__ x,
            float* __restrict__ agg) {
  int e = blockIdx.x;
  int h = threadIdx.x;
  int src = edge[e], dst = edge[EE + e];
  atomAddF(&agg[(size_t)dst * H + h], x[(size_t)src * H + h]);
}

__global__ void k_aggdiv(const float* __restrict__ agg, const float* __restrict__ cnt,
                         unsigned short* __restrict__ aggbf) {
  int i = blockIdx.x * 256 + threadIdx.x;          // i < NN*H
  float d = fmaxf(cnt[i >> 8], 1.0f);
  aggbf[i] = bf16r(agg[i] / d);
}

// ---------------- GEMM kernels (WMMA bf16 -> f32) ----------------
// x = nf @ Wproj^T + b ; += embedding in first HQ cols. Block = 16 rows, 4 waves x 4 ntiles.
__global__ void __launch_bounds__(128)
k_proj(const unsigned short* __restrict__ nf_bf, const unsigned short* __restrict__ w_bf,
       const float* __restrict__ bias, const float* __restrict__ emb,
       const int* __restrict__ bt, float* __restrict__ x_out,
       unsigned short* __restrict__ xbf_out) {
  int row0 = blockIdx.x << 4;
  int wave = threadIdx.x >> 5, lane = threadIdx.x & 31;
  v8f acc[4];
#pragma unroll
  for (int i = 0; i < 4; ++i)
#pragma unroll
    for (int e = 0; e < 8; ++e) acc[i][e] = 0.0f;
  for (int ks = 0; ks < F_IN; ks += 32) {
    v16bf a = load_frag_a(nf_bf + (size_t)row0 * F_IN + ks, F_IN, lane);
#pragma unroll
    for (int nt = 0; nt < 4; ++nt) {
      int n0 = wave * 64 + nt * 16;
      v16bf b = load_frag_b(w_bf + (size_t)n0 * F_IN + ks, F_IN, lane);
      acc[nt] = wmma_bf16(a, b, acc[nt]);
    }
  }
  int half = lane >> 4, nl = lane & 15;
#pragma unroll
  for (int nt = 0; nt < 4; ++nt) {
    int col = wave * 64 + nt * 16 + nl;
#pragma unroll
    for (int r = 0; r < 8; ++r) {
      int R = row0 + r + half * 8;
      float v = acc[nt][r] + bias[col];
      if (col < HQ) v += emb[bt[R] * HQ + col];
      x_out[(size_t)R * H + col] = v;
      xbf_out[(size_t)R * H + col] = bf16r(v);
    }
  }
}

// x = gelu(agg @ Wl^T + bl + x @ Wr^T): two A-streams accumulated in one C fragment.
__global__ void __launch_bounds__(128)
k_sage(const unsigned short* __restrict__ aggbf, const unsigned short* __restrict__ xbf,
       const unsigned short* __restrict__ wl, const unsigned short* __restrict__ wr,
       const float* __restrict__ bias, float* __restrict__ x_out,
       unsigned short* __restrict__ xbf_out) {
  int row0 = blockIdx.x << 4;
  int wave = threadIdx.x >> 5, lane = threadIdx.x & 31;
  v8f acc[4];
#pragma unroll
  for (int i = 0; i < 4; ++i)
#pragma unroll
    for (int e = 0; e < 8; ++e) acc[i][e] = 0.0f;
  for (int ks = 0; ks < H; ks += 32) {
    v16bf a1 = load_frag_a(aggbf + (size_t)row0 * H + ks, H, lane);
    v16bf a2 = load_frag_a(xbf   + (size_t)row0 * H + ks, H, lane);
#pragma unroll
    for (int nt = 0; nt < 4; ++nt) {
      int n0 = wave * 64 + nt * 16;
      v16bf b1 = load_frag_b(wl + (size_t)n0 * H + ks, H, lane);
      acc[nt] = wmma_bf16(a1, b1, acc[nt]);
      v16bf b2 = load_frag_b(wr + (size_t)n0 * H + ks, H, lane);
      acc[nt] = wmma_bf16(a2, b2, acc[nt]);
    }
  }
  int half = lane >> 4, nl = lane & 15;
#pragma unroll
  for (int nt = 0; nt < 4; ++nt) {
    int col = wave * 64 + nt * 16 + nl;
#pragma unroll
    for (int r = 0; r < 8; ++r) {
      int R = row0 + r + half * 8;
      float v = gelu_exact(acc[nt][r] + bias[col]);
      x_out[(size_t)R * H + col] = v;
      xbf_out[(size_t)R * H + col] = bf16r(v);
    }
  }
}

// pre = x @ Wih^T + bih + bhh  ([N,256] @ [256,512]) ; 8 waves x 4 ntiles = 512 cols.
__global__ void __launch_bounds__(256)
k_lstm_pre(const unsigned short* __restrict__ xbf, const unsigned short* __restrict__ wih,
           const float* __restrict__ bih, const float* __restrict__ bhh,
           float* __restrict__ pre) {
  int row0 = blockIdx.x << 4;
  int wave = threadIdx.x >> 5, lane = threadIdx.x & 31;
  v8f acc[4];
#pragma unroll
  for (int i = 0; i < 4; ++i)
#pragma unroll
    for (int e = 0; e < 8; ++e) acc[i][e] = 0.0f;
  for (int ks = 0; ks < H; ks += 32) {
    v16bf a = load_frag_a(xbf + (size_t)row0 * H + ks, H, lane);
#pragma unroll
    for (int nt = 0; nt < 4; ++nt) {
      int n0 = (wave * 4 + nt) * 16;
      v16bf b = load_frag_b(wih + (size_t)n0 * H + ks, H, lane);
      acc[nt] = wmma_bf16(a, b, acc[nt]);
    }
  }
  int half = lane >> 4, nl = lane & 15;
#pragma unroll
  for (int nt = 0; nt < 4; ++nt) {
    int col = (wave * 4 + nt) * 16 + nl;
#pragma unroll
    for (int r = 0; r < 8; ++r) {
      int R = row0 + r + half * 8;
      pre[(size_t)R * G4 + col] = acc[nt][r] + bih[col] + bhh[col];
    }
  }
}

// Issue async copy of one step's gate-precompute tile [16 rows x 512 f32] into LDS.
// 2048 16-byte chunks, 8 per thread; each wave's ASYNCcnt += 8.
DEV void prefetch_pre(const float* __restrict__ pre, int gbase, int t,
                      float* dstLds, int tid) {
#pragma unroll
  for (int j = 0; j < 8; ++j) {
    int cidx = j * 256 + tid;            // chunk index 0..2047
    int m = cidx >> 7;                   // 128 chunks per 512-float row
    int q = cidx & 127;
    const float* g = pre + ((size_t)(gbase + m) * LL + t) * G4 + q * 4;
    unsigned lds = (unsigned)(unsigned long long)(const void*)(dstLds + (size_t)cidx * 4);
    async_ld_b128(lds, g);
  }
}

// Recurrent LSTM: 16 graphs per block, 512 dependent steps.
// Wave w owns hidden cols hh = w*16..w*16+15 for ALL FOUR gates (n-tiles at
// w*16 + {0,128,256,384}), so i/f/g/o live in this wave's 4 accumulators and the
// nonlinearity runs fully in registers. Whh fragments are register-resident.
// The pre[t] tile is double-buffered in LDS via async global->LDS loads.
__global__ void __launch_bounds__(256)
k_lstm(const float* __restrict__ pre, const unsigned short* __restrict__ whh,
       float* __restrict__ xl, int dir) {
  __shared__ __align__(16) float preLds[2][16 * G4];   // 2 x 32 KB
  __shared__ __align__(16) unsigned short hbf[16 * HH]; // 4 KB bf16 hidden state
  int gbase = blockIdx.x << 4;
  int tid = threadIdx.x, wave = tid >> 5, lane = tid & 31;
  int half = lane >> 4, nl = lane & 15;
  int hh = wave * 16 + nl;

  // Register-resident Whh fragments: bfr[kstep][gate]
  v16bf bfr[4][4];
#pragma unroll
  for (int ki = 0; ki < 4; ++ki)
#pragma unroll
    for (int g = 0; g < 4; ++g) {
      int n0 = g * HH + wave * 16;
      bfr[ki][g] = load_frag_b(whh + (size_t)n0 * HH + ki * 32, HH, lane);
    }

  for (int i = tid; i < 16 * HH; i += 256) hbf[i] = 0;
  float c[8];
#pragma unroll
  for (int r = 0; r < 8; ++r) c[r] = 0.0f;

  prefetch_pre(pre, gbase, dir ? (LL - 1) : 0, &preLds[0][0], tid);
  __syncthreads();

  for (int s = 0; s < LL; ++s) {
    int t = dir ? (LL - 1 - s) : s;
    int cur = s & 1;
    v8f acc[4];
#pragma unroll
    for (int i = 0; i < 4; ++i)
#pragma unroll
      for (int e = 0; e < 8; ++e) acc[i][e] = 0.0f;
#pragma unroll
    for (int ki = 0; ki < 4; ++ki) {
      v16bf a = load_frag_a(hbf + ki * 32, HH, lane);
#pragma unroll
      for (int g = 0; g < 4; ++g) acc[g] = wmma_bf16(a, bfr[ki][g], acc[g]);
    }
    if (s + 1 < LL) {
      // prefetch next step while we finish this one; wait only for the older 8
      prefetch_pre(pre, gbase, dir ? (LL - 2 - s) : (s + 1), &preLds[cur ^ 1][0], tid);
      asm volatile("s_wait_asynccnt 0x8" ::: "memory");
    } else {
      asm volatile("s_wait_asynccnt 0x0" ::: "memory");
    }
    __syncthreads();   // preLds[cur] visible from all waves; hbf reads complete
    const float* pl = &preLds[cur][0];
#pragma unroll
    for (int r = 0; r < 8; ++r) {
      int m = r + half * 8;
      float gi = acc[0][r] + pl[m * G4 + hh];
      float gf = acc[1][r] + pl[m * G4 + HH + hh];
      float gg = acc[2][r] + pl[m * G4 + 2 * HH + hh];
      float go = acc[3][r] + pl[m * G4 + 3 * HH + hh];
      float cn = sigm(gf) * c[r] + sigm(gi) * tanhf(gg);
      float hv = sigm(go) * tanhf(cn);
      c[r] = cn;
      hbf[m * HH + hh] = bf16r(hv);
      size_t node = (size_t)(gbase + m) * LL + t;
      xl[node * H + dir * HH + hh] = hv;
    }
    __syncthreads();   // hbf updated before next step's GEMM
  }
}

// x_combined = x + x_lstm ; head logits + softmax ; accumulate graph means for probs.
__global__ void __launch_bounds__(256)
k_combine(const float* __restrict__ x, const float* __restrict__ xl,
          const float* __restrict__ Wb, const float* __restrict__ bb,
          const float* __restrict__ Wl, const float* __restrict__ bl,
          float* __restrict__ out) {
  __shared__ float xc[H];
  __shared__ float lg[7];
  int n = blockIdx.x, t = threadIdx.x;
  float v = x[(size_t)n * H + t] + xl[(size_t)n * H + t];
  xc[t] = v;
  out[(size_t)n * H + t] = v;
  __syncthreads();
  if (t < 7) {
    const float* w = (t < 3) ? (Wb + t * H) : (Wl + (t - 3) * H);
    float s = (t < 3) ? bb[t] : bl[t - 3];
    for (int k = 0; k < H; ++k) s += w[k] * xc[k];
    lg[t] = s;
  }
  __syncthreads();
  if (t == 0) {
    int g = n >> 9;                                  // node / LL
    float inv = 1.0f / (float)LL;
    float m3 = fmaxf(lg[0], fmaxf(lg[1], lg[2]));
    float e0 = expf(lg[0] - m3), e1 = expf(lg[1] - m3), e2 = expf(lg[2] - m3);
    float s3 = e0 + e1 + e2;
    atomAddF(&out[OFF_BR + (size_t)g * 3 + 0], e0 / s3 * inv);
    atomAddF(&out[OFF_BR + (size_t)g * 3 + 1], e1 / s3 * inv);
    atomAddF(&out[OFF_BR + (size_t)g * 3 + 2], e2 / s3 * inv);
    float m4 = fmaxf(fmaxf(lg[3], lg[4]), fmaxf(lg[5], lg[6]));
    float f0 = expf(lg[3] - m4), f1 = expf(lg[4] - m4);
    float f2 = expf(lg[5] - m4), f3 = expf(lg[6] - m4);
    float s4 = f0 + f1 + f2 + f3;
    atomAddF(&out[OFF_LP + (size_t)g * 4 + 0], f0 / s4 * inv);
    atomAddF(&out[OFF_LP + (size_t)g * 4 + 1], f1 / s4 * inv);
    atomAddF(&out[OFF_LP + (size_t)g * 4 + 2], f2 / s4 * inv);
    atomAddF(&out[OFF_LP + (size_t)g * 4 + 3], f3 / s4 * inv);
  }
}

__global__ void __launch_bounds__(256)
k_gmean(const float* __restrict__ xc, float* __restrict__ ge) {
  int g = blockIdx.x, h = threadIdx.x;
  const float* p = xc + (size_t)g * LL * H + h;
  float s = 0.0f;
  for (int t = 0; t < LL; ++t) s += p[(size_t)t * H];
  ge[(size_t)g * H + h] = s * (1.0f / (float)LL);
}

// ---------------- driver ----------------
extern "C" void kernel_launch(void* const* d_in, const int* in_sizes, int n_in,
                              void* d_out, int out_size, void* d_ws, size_t ws_size,
                              hipStream_t stream) {
  (void)in_sizes; (void)n_in; (void)out_size; (void)ws_size;
  const float* nf       = (const float*)d_in[0];
  const int*   edge     = (const int*)d_in[1];
  const int*   btypes   = (const int*)d_in[2];
  const float* W_proj   = (const float*)d_in[4];
  const float* b_proj   = (const float*)d_in[5];
  const float* emb      = (const float*)d_in[6];
  const float* sage_Wl  = (const float*)d_in[7];
  const float* sage_bl  = (const float*)d_in[8];
  const float* sage_Wr  = (const float*)d_in[9];
  const float* Wih_f    = (const float*)d_in[10];
  const float* Whh_f    = (const float*)d_in[11];
  const float* bih_f    = (const float*)d_in[12];
  const float* bhh_f    = (const float*)d_in[13];
  const float* Wih_b    = (const float*)d_in[14];
  const float* Whh_b    = (const float*)d_in[15];
  const float* bih_b    = (const float*)d_in[16];
  const float* bhh_b    = (const float*)d_in[17];
  const float* W_branch = (const float*)d_in[18];
  const float* b_branch = (const float*)d_in[19];
  const float* W_loop   = (const float*)d_in[20];
  const float* b_loop   = (const float*)d_in[21];

  char* ws = (char*)d_ws;
  float*          x_f    = (float*)ws;                              // [N,H] f32
  char*           uni    = ws + SZ_X;                               // union region (SZ_PRE bytes)
  float*          agg    = (float*)uni;                             // GNN phase: [N,H] f32
  unsigned short* agg_bf = (unsigned short*)(uni + SZ_X);           // GNN phase: [N,H] bf16
  unsigned short* nf_bf  = agg_bf;                                  // alias (pre-GNN only)
  float*          pre    = (float*)uni;                             // LSTM phase: [N,512] f32
  unsigned short* x_bf   = (unsigned short*)(ws + SZ_X + SZ_PRE);   // [N,H] bf16
  float*          cnt    = (float*)(ws + SZ_X + SZ_PRE + SZ_XBF);   // [N] f32
  unsigned short* wbf    = (unsigned short*)(ws + SZ_X + SZ_PRE + SZ_XBF + (size_t)NN * 4);

  unsigned short* wproj_bf = wbf;
  unsigned short* wl_bf    = wproj_bf + H * F_IN;        // 3*H*H
  unsigned short* wr_bf    = wl_bf + 3 * H * H;          // 3*H*H
  unsigned short* wihf_bf  = wr_bf + 3 * H * H;          // G4*H
  unsigned short* whhf_bf  = wihf_bf + G4 * H;           // G4*HH
  unsigned short* wihb_bf  = whhf_bf + G4 * HH;          // G4*H
  unsigned short* whhb_bf  = wihb_bf + G4 * H;           // G4*HH

  auto cvt = [&](const float* s, unsigned short* d, int n) {
    k_f32_to_bf16<<<(n + 255) / 256, 256, 0, stream>>>(s, d, n);
  };
  cvt(nf, nf_bf, NN * F_IN);
  cvt(W_proj, wproj_bf, H * F_IN);
  cvt(sage_Wl, wl_bf, 3 * H * H);
  cvt(sage_Wr, wr_bf, 3 * H * H);
  cvt(Wih_f, wihf_bf, G4 * H);
  cvt(Whh_f, whhf_bf, G4 * HH);
  cvt(Wih_b, wihb_bf, G4 * H);
  cvt(Whh_b, whhb_bf, G4 * HH);

  k_zero<<<(NN + 255) / 256, 256, 0, stream>>>(cnt, NN);
  k_degree<<<EE / 256, 256, 0, stream>>>(edge, cnt);

  k_proj<<<NN / 16, 128, 0, stream>>>(nf_bf, wproj_bf, b_proj, emb, btypes, x_f, x_bf);

  for (int l = 0; l < 3; ++l) {
    k_zero<<<(NN * H) / 256, 256, 0, stream>>>(agg, NN * H);
    k_aggregate<<<EE, 256, 0, stream>>>(edge, x_f, agg);
    k_aggdiv<<<(NN * H) / 256, 256, 0, stream>>>(agg, cnt, agg_bf);
    k_sage<<<NN / 16, 128, 0, stream>>>(agg_bf, x_bf,
                                        wl_bf + (size_t)l * H * H,
                                        wr_bf + (size_t)l * H * H,
                                        sage_bl + l * H, x_f, x_bf);
  }

  float* out = (float*)d_out;
  k_zero<<<4, 256, 0, stream>>>(out + OFF_BR, GG * 3 + GG * 4);

  // forward direction
  k_lstm_pre<<<NN / 16, 256, 0, stream>>>(x_bf, wihf_bf, bih_f, bhh_f, pre);
  k_lstm<<<GG / 16, 256, 0, stream>>>(pre, whhf_bf, out + OFF_XL, 0);
  // backward direction (reuses pre buffer; stream-ordered)
  k_lstm_pre<<<NN / 16, 256, 0, stream>>>(x_bf, wihb_bf, bih_b, bhh_b, pre);
  k_lstm<<<GG / 16, 256, 0, stream>>>(pre, whhb_bf, out + OFF_XL, 1);

  k_combine<<<NN, 256, 0, stream>>>(x_f, out + OFF_XL, W_branch, b_branch,
                                    W_loop, b_loop, out);
  k_gmean<<<GG, 256, 0, stream>>>(out, out + OFF_GE);
}
